// GatherModel_63101659513266
// MI455X (gfx1250) — compile-verified, matrix operand order checked by reading, not run on hardware
//
#include <hip/hip_runtime.h>
#include <hip/hip_bf16.h>

#define D     49
#define DIN   56
#define HP    64          // padded hidden stride
#define NLAY  3
#define NGR   512
#define BN_EPS 1e-5f
#define STATS_BLOCKS 120
#define ROWS_PER_BLOCK 256

typedef __attribute__((ext_vector_type(16))) __bf16    v16bf;
typedef __attribute__((ext_vector_type(8)))  float     v8f;
typedef __attribute__((ext_vector_type(4)))  float     v4f;
typedef __attribute__((ext_vector_type(2)))  float     v2f;
typedef __attribute__((ext_vector_type(8)))  unsigned  v8u;

__device__ __forceinline__ unsigned short f32_to_bf16(float f) {
  unsigned u = __builtin_bit_cast(unsigned, f);
  u += 0x7FFFu + ((u >> 16) & 1u);            // round-to-nearest-even
  return (unsigned short)(u >> 16);
}
__device__ __forceinline__ unsigned packbf(float lo, float hi) {
  return (unsigned)f32_to_bf16(lo) | ((unsigned)f32_to_bf16(hi) << 16);
}

// ---- convert + pad all weights to bf16, stored n-major [slot][n*64+k] ----
__global__ void prep_weights(const float* __restrict__ W0,
                             const float* __restrict__ W1,
                             const float* __restrict__ W2,
                             unsigned short* __restrict__ wg) {
  const int per = HP * HP;
  for (int idx = blockIdx.x * blockDim.x + threadIdx.x; idx < 7 * per;
       idx += gridDim.x * blockDim.x) {
    int slot = idx / per;
    int r = idx % per;
    int n = r / HP, k = r % HP;
    float v = 0.f;
    if (n < D) {
      if (slot == 0)      { if (k < DIN) v = W0[k * D + n]; }
      else if (slot < 4)  { int l = slot - 1; if (k < D) v = W1[(l * D + k) * D + n]; }
      else                { int l = slot - 4; if (k < D) v = W2[(l * D + k) * D + n]; }
    }
    wg[idx] = f32_to_bf16(v);
  }
}

// ---- WMMA GEMM: out[N(pad),64] = A[N,kvalid(<=64)] @ W(64x64 bf16, padded) + bias
// 256 threads = 8 waves; each wave: TWO 16-row M tiles x full 64-col N.
// All 8 B-fragments hoisted to registers; A staged as packed bf16 in LDS
// (stride 72 bf16 -> 16B-aligned conflict-free b128 reads); weight panel copied
// raw via async-to-LDS (ASYNCcnt). Output buffer is padded: unguarded stores.
__global__ __launch_bounds__(256)
void gemm_bias(const float* __restrict__ A, int strideA, int kvalid, int Nn,
               const unsigned short* __restrict__ wg,
               const float* __restrict__ bias,
               float* __restrict__ out) {
  __shared__ unsigned short Asb[ROWS_PER_BLOCK * 72];
  __shared__ unsigned short Wsb[HP * 72];
  unsigned* As32 = (unsigned*)Asb;
  const unsigned* Ws32 = (const unsigned*)Wsb;
  const int tid = threadIdx.x;
  const int rowBase = blockIdx.x * ROWS_PER_BLOCK;

  // --- async raw copy of bf16 weight panel (8KB) into padded LDS, 16B chunks ---
  {
    unsigned wbase = (unsigned)(unsigned long long)(void*)Wsb;  // LDS byte offset
    for (int i = tid; i < HP * 8; i += 256) {
      int n = i >> 3, c = i & 7;
      unsigned ldsOff = wbase + (unsigned)(n * 144 + c * 16);
      unsigned long long gaddr = (unsigned long long)(wg + n * HP + c * 8);
      asm volatile("global_load_async_to_lds_b128 %0, %1, off"
                   :: "v"(ldsOff), "v"(gaddr) : "memory");
    }
  }

  // --- stage A tile: f32 -> packed bf16 pairs (one dword per pair) ---
  for (int i = tid; i < ROWS_PER_BLOCK * 32; i += 256) {
    int r = i >> 5, cp = i & 31;
    int row = rowBase + r;
    int c = cp * 2;
    float v0 = 0.f, v1 = 0.f;
    if (row < Nn) {
      const float* ar = A + (long long)row * strideA;
      if (c < kvalid)     v0 = ar[c];
      if (c + 1 < kvalid) v1 = ar[c + 1];
    }
    As32[r * 36 + cp] = packbf(v0, v1);
  }
  asm volatile("s_wait_asynccnt 0" ::: "memory");
  __syncthreads();

  const int wave = tid >> 5;
  const int lane = tid & 31;
  const int half = lane >> 4;                 // K-swizzle half
  const int l16  = lane & 15;
  const int m0   = wave * 32;                 // two 16-row tiles per wave

  // --- hoist all B fragments (4 N-tiles x 2 K-steps) into registers ---
  v16bf bfrag[8];
  #pragma unroll
  for (int ks2 = 0; ks2 < 2; ++ks2) {
    const int koff = ks2 * 32 + half * 8;
    #pragma unroll
    for (int t = 0; t < 4; ++t) {
      const unsigned* wp = &Ws32[(t * 16 + l16) * 36 + (koff >> 1)];
      v8u bu;
      #pragma unroll
      for (int i = 0; i < 4; ++i) { bu[i] = wp[i]; bu[4 + i] = wp[8 + i]; }
      bfrag[ks2 * 4 + t] = __builtin_bit_cast(v16bf, bu);
    }
  }

  v8f acc[2][4];
  #pragma unroll
  for (int t = 0; t < 4; ++t) {
    int col = t * 16 + l16;
    float b = (col < D) ? bias[col] : 0.f;
    #pragma unroll
    for (int r = 0; r < 8; ++r) { acc[0][t][r] = b; acc[1][t][r] = b; }
  }

  #pragma unroll
  for (int mt = 0; mt < 2; ++mt) {
    #pragma unroll
    for (int ks2 = 0; ks2 < 2; ++ks2) {
      const int koff = ks2 * 32 + half * 8;
      const unsigned* ap = &As32[(m0 + mt * 16 + l16) * 36 + (koff >> 1)];
      v8u au;
      #pragma unroll
      for (int i = 0; i < 4; ++i) { au[i] = ap[i]; au[4 + i] = ap[8 + i]; }
      v16bf a = __builtin_bit_cast(v16bf, au);
      #pragma unroll
      for (int t = 0; t < 4; ++t) {
        acc[mt][t] = __builtin_amdgcn_wmma_f32_16x16x32_bf16(
            false, a, false, bfrag[ks2 * 4 + t], (short)0, acc[mt][t], false, false);
      }
    }
  }

  // --- unguarded stores (out buffer padded to ROWS_PER_BLOCK multiple) ---
  // C/D layout: VGPR r -> M = half*8 + r ; N = t*16 + l16
  #pragma unroll
  for (int mt = 0; mt < 2; ++mt) {
    float* op = out + (long long)(rowBase + m0 + mt * 16 + half * 8) * HP + l16;
    #pragma unroll
    for (int t = 0; t < 4; ++t) {
      #pragma unroll
      for (int r = 0; r < 8; ++r) op[r * HP + t * 16] = acc[mt][t][r];
    }
  }
}

// ---- z = (1+eps[l]) * h  (vec4) ----
__global__ void scale_self(const float* __restrict__ h, float* __restrict__ z,
                           const float* __restrict__ eps, int l, long long total4) {
  long long i = (long long)blockIdx.x * blockDim.x + threadIdx.x;
  if (i < total4) {
    float s = 1.f + eps[l];
    v4f v = ((const v4f*)h)[i];
    v4f r; r.x = s * v.x; r.y = s * v.y; r.z = s * v.z; r.w = s * v.w;
    ((v4f*)z)[i] = r;
  }
}

// ---- z[dst] += h[src]: 32 lanes/edge, float2 loads, L2 f32 atomics ----
__global__ void edge_agg(const float* __restrict__ h, float* __restrict__ z,
                         const int* __restrict__ src, const int* __restrict__ dst,
                         long long E) {
  long long g = (long long)blockIdx.x * blockDim.x + threadIdx.x;
  long long e = g >> 5;
  int c = (int)(g & 31) * 2;
  if (e < E && c < D) {
    int s = src[e], d = dst[e];
    v2f v = *(const v2f*)(h + (long long)s * HP + c);
    float* zp = z + (long long)d * HP + c;
    unsafeAtomicAdd(zp, v.x);
    if (c + 1 < D) unsafeAtomicAdd(zp + 1, v.y);
  }
}

// ---- BN phase 1: per-block column partial sums / sumsq ----
__global__ __launch_bounds__(256)
void bn_stats(const float* __restrict__ t, int Nn,
              float* __restrict__ partS, float* __restrict__ partQ) {
  __shared__ float sS[256], sQ[256];
  int tid = threadIdx.x;
  int c = tid & 63;
  int sub = tid >> 6;
  float s = 0.f, q = 0.f;
  for (int r = blockIdx.x * 4 + sub; r < Nn; r += gridDim.x * 4) {
    float v = t[(long long)r * HP + c];
    s += v; q += v * v;
  }
  sS[tid] = s; sQ[tid] = q;
  __syncthreads();
  if (sub == 0) {
    partS[blockIdx.x * 64 + c] = sS[c] + sS[64 + c] + sS[128 + c] + sS[192 + c];
    partQ[blockIdx.x * 64 + c] = sQ[c] + sQ[64 + c] + sQ[128 + c] + sQ[192 + c];
  }
}

// ---- BN phase 2 (1 block, deterministic): fused scale/shift per column ----
__global__ void bn_finalize(const float* __restrict__ partS, const float* __restrict__ partQ,
                            int nblk, int Nn,
                            const float* __restrict__ gamma, const float* __restrict__ beta,
                            float* __restrict__ scale, float* __restrict__ shift) {
  int c = threadIdx.x;
  float s = 0.f, q = 0.f;
  for (int b = 0; b < nblk; ++b) { s += partS[b * 64 + c]; q += partQ[b * 64 + c]; }
  float inv = 1.f / (float)Nn;
  float mu  = s * inv;
  float var = fmaxf(q * inv - mu * mu, 0.f);
  float sc = 0.f, sh = 0.f;
  if (c < D) {
    sc = gamma[c] * rsqrtf(var + BN_EPS);
    sh = beta[c] - mu * sc;
  }
  scale[c] = sc; shift[c] = sh;
}

// ---- out = relu(in*scale + shift) (vec4); padded columns stay exactly 0 ----
__global__ void bn_relu(const float* __restrict__ in, float* __restrict__ out,
                        const float* __restrict__ scale, const float* __restrict__ shift,
                        long long total4) {
  long long i = (long long)blockIdx.x * blockDim.x + threadIdx.x;
  if (i < total4) {
    int c = (int)((i & 15) * 4);
    v4f v = ((const v4f*)in)[i];
    v4f r;
    r.x = fmaxf(fmaf(v.x, scale[c + 0], shift[c + 0]), 0.f);
    r.y = fmaxf(fmaf(v.y, scale[c + 1], shift[c + 1]), 0.f);
    r.z = fmaxf(fmaf(v.z, scale[c + 2], shift[c + 2]), 0.f);
    r.w = fmaxf(fmaf(v.w, scale[c + 3], shift[c + 3]), 0.f);
    ((v4f*)out)[i] = r;
  }
}

__global__ void zero_pool(float* __restrict__ pool, float* __restrict__ cnt) {
  int i = blockIdx.x * blockDim.x + threadIdx.x;
  if (i < NGR * HP) pool[i] = 0.f;
  if (i < NGR)      cnt[i]  = 0.f;
}

__global__ void pool_acc(const float* __restrict__ h, const int* __restrict__ batch,
                         float* __restrict__ pool, float* __restrict__ cnt, int Nn) {
  long long g = (long long)blockIdx.x * blockDim.x + threadIdx.x;
  int n = (int)(g >> 6);
  int c = (int)(g & 63);
  if (n < Nn) {
    int b = batch[n];
    if (c < D)  unsafeAtomicAdd(&pool[b * HP + c], h[(long long)n * HP + c]);
    if (c == 0) unsafeAtomicAdd(&cnt[b], 1.f);
  }
}

__global__ void pool_div(const float* __restrict__ pool, const float* __restrict__ cnt,
                         float* __restrict__ out) {
  int i = blockIdx.x * blockDim.x + threadIdx.x;
  if (i < NGR * D) {
    int g = i / D, c = i % D;
    out[i] = pool[g * HP + c] / fmaxf(cnt[g], 1.f);
  }
}

extern "C" void kernel_launch(void* const* d_in, const int* in_sizes, int n_in,
                              void* d_out, int out_size, void* d_ws, size_t ws_size,
                              hipStream_t stream) {
  const float* x   = (const float*)d_in[0];
  const float* W0  = (const float*)d_in[1];
  const float* b0  = (const float*)d_in[2];
  const float* W1  = (const float*)d_in[3];
  const float* b1  = (const float*)d_in[4];
  const float* g1  = (const float*)d_in[5];
  const float* be1 = (const float*)d_in[6];
  const float* W2  = (const float*)d_in[7];
  const float* b2  = (const float*)d_in[8];
  const float* g2  = (const float*)d_in[9];
  const float* be2 = (const float*)d_in[10];
  const float* eps = (const float*)d_in[11];
  const int*   ei  = (const int*)d_in[12];
  const int*   bat = (const int*)d_in[13];

  const int Nn = in_sizes[0] / DIN;
  const long long E = in_sizes[12] / 2;
  const int* src = ei;
  const int* dst = ei + E;

  const int gblk = (Nn + ROWS_PER_BLOCK - 1) / ROWS_PER_BLOCK;
  const int NnPad = gblk * ROWS_PER_BLOCK;    // padded rows: unguarded GEMM stores

  char* p = (char*)d_ws;
  auto alloc = [&](size_t bytes) -> void* {
    void* r = (void*)p;
    p += (bytes + 255) & ~(size_t)255;
    return r;
  };
  float* h  = (float*)alloc((size_t)NnPad * HP * 4);
  float* z  = (float*)alloc((size_t)NnPad * HP * 4);
  float* tb = (float*)alloc((size_t)NnPad * HP * 4);
  unsigned short* wg = (unsigned short*)alloc((size_t)7 * HP * HP * 2);
  float* partS = (float*)alloc((size_t)STATS_BLOCKS * 64 * 4);
  float* partQ = (float*)alloc((size_t)STATS_BLOCKS * 64 * 4);
  float* scal  = (float*)alloc(64 * 4);
  float* shft  = (float*)alloc(64 * 4);
  float* pool  = (float*)alloc((size_t)NGR * HP * 4);
  float* cnt   = (float*)alloc((size_t)NGR * 4);

  const long long totalNH = (long long)Nn * HP;
  const long long total4  = totalNH / 4;
  const int nb4  = (int)((total4 + 255) / 256);
  const int eb   = (int)((E * 32 + 255) / 256);

  prep_weights<<<112, 256, 0, stream>>>(W0, W1, W2, wg);

  // lin0: h = x @ W0 + b0
  gemm_bias<<<gblk, 256, 0, stream>>>(x, DIN, DIN, Nn, wg, b0, h);

  for (int l = 0; l < NLAY; ++l) {
    scale_self<<<nb4, 256, 0, stream>>>(h, z, eps, l, total4);
    edge_agg<<<eb, 256, 0, stream>>>(h, z, src, dst, E);

    gemm_bias<<<gblk, 256, 0, stream>>>(z, HP, HP, Nn, wg + (size_t)(1 + l) * HP * HP,
                                        b1 + l * D, tb);
    bn_stats<<<STATS_BLOCKS, 256, 0, stream>>>(tb, Nn, partS, partQ);
    bn_finalize<<<1, 64, 0, stream>>>(partS, partQ, STATS_BLOCKS, Nn,
                                      g1 + l * D, be1 + l * D, scal, shft);
    bn_relu<<<nb4, 256, 0, stream>>>(tb, tb, scal, shft, total4);

    gemm_bias<<<gblk, 256, 0, stream>>>(tb, HP, HP, Nn, wg + (size_t)(4 + l) * HP * HP,
                                        b2 + l * D, z);
    bn_stats<<<STATS_BLOCKS, 256, 0, stream>>>(z, Nn, partS, partQ);
    bn_finalize<<<1, 64, 0, stream>>>(partS, partQ, STATS_BLOCKS, Nn,
                                      g2 + l * D, be2 + l * D, scal, shft);
    bn_relu<<<nb4, 256, 0, stream>>>(z, h, scal, shft, total4);
  }

  zero_pool<<<(NGR * HP + 255) / 256, 256, 0, stream>>>(pool, cnt);
  pool_acc<<<(int)((totalNH + 255) / 256), 256, 0, stream>>>(h, bat, pool, cnt, Nn);
  pool_div<<<(NGR * D + 255) / 256, 256, 0, stream>>>(pool, cnt, (float*)d_out);
}
